// Model_14731737825320
// MI455X (gfx1250) — matrix operand order repaired and checked by reading, not compiled, not run on hardware
//
#include <hip/hip_runtime.h>
#include <math.h>

// ---------------------------------------------------------------------------
// Change-point transformer for MI455X (gfx1250), wave32 + WMMA + async-LDS.
// All GEMMs run through one generic f16 WMMA kernel (v_wmma_f32_16x16x32_f16,
// f32 accumulate) whose tiles are staged global->LDS with
// GLOBAL_LOAD_ASYNC_TO_LDS_B128 (ASYNCcnt / s_wait_asynccnt, ISA 08).
// The data-dependent change point `cp` is computed on device and consumed by
// downstream kernels (graph-capture safe: fixed launch shapes).
//
// Assumed d_in order (insertion order of setup_inputs() and nested params):
//  0 x_enc(f32 8*512*32) 1 x_mark_enc(i32 8*512*4) 2 x_dec 3 x_mark_dec
//  4 glob_conv(3*32*512) 5 grp_conv0(3*16*512) 6 grp_conv1
//  7..14  time_attn: wq wk wv wo bq bk bv bo
// 15..22  loc_attn0: wq wk wv wo bq bk bv bo
// 23..30  loc_attn1: wq wk wv wo bq bk bv bo
// 31 norm1_g 32 norm1_b 33 norm2_g 34 norm2_b
// 35 mlp_w1 36 mlp_b1 37 mlp_w2 38 mlp_b2 39 proj_w 40 proj_b
// ---------------------------------------------------------------------------

typedef __attribute__((ext_vector_type(16))) _Float16 v16h;
typedef __attribute__((ext_vector_type(8)))  float    v8f;

#define B_    8
#define L_    512
#define C_    32
#define D_    512
#define H_    8
#define DK_   64
#define DFF_  2048
#define PRED_ 96

#define BM 64
#define BN 64
#define BK 32

union F16Frag { v16h v; unsigned u[8]; _Float16 h[16]; };

// Async global->LDS copy, 16 bytes per lane (GLOBAL_LOAD_ASYNC_TO_LDS_B128,
// GV mode: 64-bit vaddr, dest = per-lane LDS byte address in a VGPR).
// For LDS-aperture flat pointers, addr[31:0] is the LDS byte offset (ISA 10.2).
__device__ __forceinline__ void async_copy_b128(unsigned lds_addr, const void* gaddr) {
  asm volatile("global_load_async_to_lds_b128 %0, %1, off"
               :: "v"(lds_addr), "v"(gaddr) : "memory");
}
__device__ __forceinline__ void wait_asynccnt0() {
  asm volatile("s_wait_asynccnt 0" ::: "memory");
}

// ---------------------------------------------------------------------------
// Generic NT WMMA GEMM:  C[M,N] = A[M,K] * Bt[N,K]^T (+bias, act, f16/transposed
// outputs). Batched via grid.z: bz -> (b = bz/Hdiv, h = bz%Hdiv) with separate
// b/h strides so Q/K/V head slices and per-(b,h) score blocks address directly.
// Tile: 64x64x32 per workgroup (8 waves, 2 x 16x16 tiles per wave).
// ---------------------------------------------------------------------------
__global__ __launch_bounds__(256) void k_gemm(
    const _Float16* __restrict__ A,  long long sAb,  long long sAh,  int lda,
    const _Float16* __restrict__ Bt, long long sBb,  long long sBh,  int ldbt,
    float* __restrict__ C,           long long sCb,  long long sCh,  int ldc,
    _Float16* __restrict__ C16,      long long sC16b,long long sC16h,int ldc16,
    _Float16* __restrict__ Tp,       const float* __restrict__ bias,
    int M, int N, int K, int Hdiv, int act)
{
  __shared__ __align__(16) _Float16 As[BM][BK + 8];
  __shared__ __align__(16) _Float16 Bs[BN][BK + 8];

  const int bz = blockIdx.z;
  const int bb = bz / Hdiv, hh = bz % Hdiv;
  A  += (size_t)bb * sAb + (size_t)hh * sAh;
  Bt += (size_t)bb * sBb + (size_t)hh * sBh;

  const int m0  = blockIdx.y * BM;
  const int n0  = blockIdx.x * BN;
  const int tid = threadIdx.x;
  const int lane = tid & 31;
  const int wave = tid >> 5;
  const int lh = lane >> 4, ln = lane & 15;   // half (0/1), row-in-half
  const int wc = wave & 3,  wr = wave >> 2;   // subtile col, subtile row base

  v8f acc0, acc1;
#pragma unroll
  for (int i = 0; i < 8; ++i) { acc0[i] = 0.0f; acc1[i] = 0.0f; }

  const int lrow = tid >> 2;            // 0..63
  const int lseg = (tid & 3) << 3;      // 0,8,16,24  (f16 elements)
  const _Float16* ag = A  + (size_t)(m0 + lrow) * lda  + lseg;
  const _Float16* bg = Bt + (size_t)(n0 + lrow) * ldbt + lseg;
  const unsigned ldsA = (unsigned)(uintptr_t)&As[lrow][lseg];
  const unsigned ldsB = (unsigned)(uintptr_t)&Bs[lrow][lseg];

  for (int k0 = 0; k0 < K; k0 += BK) {
    __syncthreads();                       // previous fragment reads done
    async_copy_b128(ldsA, (const void*)(ag + k0));
    async_copy_b128(ldsB, (const void*)(bg + k0));
    if (k0 + BK < K) {
      __builtin_prefetch((const void*)(ag + k0 + BK), 0, 1);
      __builtin_prefetch((const void*)(bg + k0 + BK), 0, 1);
    }
    wait_asynccnt0();                      // own wave's async copies landed
    __syncthreads();                       // all waves' tiles visible

    F16Frag a0, a1, bf;
#pragma unroll
    for (int p = 0; p < 8; ++p) {
      // A 16x32 f16 layout: VGPR p<4: K=2p{,+1}+8*lh ; p>=4: K=2p+8{,+1}+8*lh
      const int ka = 2 * p + 8 * lh + ((p >= 4) ? 8 : 0);
      a0.u[p] = *(const unsigned*)(&As[wr * 16 + ln][ka]);
      a1.u[p] = *(const unsigned*)(&As[(wr + 2) * 16 + ln][ka]);
      // B 32x16 f16 layout: same K order as A (p<4: K=2p+8*lh; p>=4: K=2p+8+8*lh), N=ln
      const int kb = 2 * p + 8 * lh + ((p >= 4) ? 8 : 0);
      bf.u[p] = *(const unsigned*)(&Bs[wc * 16 + ln][kb]);
    }
    acc0 = __builtin_amdgcn_wmma_f32_16x16x32_f16(false, a0.v, false, bf.v,
                                                  (short)0, acc0, false, false);
    acc1 = __builtin_amdgcn_wmma_f32_16x16x32_f16(false, a1.v, false, bf.v,
                                                  (short)0, acc1, false, false);
  }

  if (C)   C   += (size_t)bb * sCb   + (size_t)hh * sCh;
  if (C16) C16 += (size_t)bb * sC16b + (size_t)hh * sC16h;

  const int nn = n0 + wc * 16 + ln;
  const float bv = bias ? bias[nn] : 0.0f;
#pragma unroll
  for (int t2 = 0; t2 < 2; ++t2) {
    const int mt = m0 + (wr + 2 * t2) * 16;
#pragma unroll
    for (int rr = 0; rr < 8; ++rr) {
      const int mm = mt + rr + 8 * lh;     // C layout: M = rr + 8*lh, N = ln
      float v = (t2 ? acc1[rr] : acc0[rr]) + bv;
      if (act == 1) v = 0.5f * v * (1.0f + erff(v * 0.70710678118654752f));
      if (C)   C[(size_t)mm * ldc + nn] = v;
      if (C16) C16[(size_t)mm * ldc16 + nn] = (_Float16)v;
      if (Tp)  // per-batch-of-L transpose: (b, n, l) layout for V^T
        Tp[(size_t)(mm >> 9) * ((size_t)N << 9) + ((size_t)nn << 9)
           + (size_t)(mm & (L_ - 1))] = (_Float16)v;
    }
  }
}

// f32 weight (K x N) -> f16 transposed (N x Kpad), zero-padded K.
__global__ void k_w_t(const float* __restrict__ src, _Float16* __restrict__ dst,
                      int K, int N, int Kpad) {
  int idx = blockIdx.x * 256 + threadIdx.x;
  if (idx >= N * Kpad) return;
  int n = idx / Kpad, k = idx % Kpad;
  dst[idx] = (k < K) ? (_Float16)src[(size_t)k * N + n] : (_Float16)0.0f;
}

// im2col for the global conv (C=32, circular padding over full L).
__global__ void k_im2col_glob(const float* __restrict__ x, _Float16* __restrict__ Xc) {
  int bl = blockIdx.x; int b = bl / L_, l = bl % L_;
  int t = threadIdx.x;
  if (t >= 96) return;
  int kk = t / C_, c = t % C_;
  int pos = l - 1 + kk;
  if (pos < 0)   pos = L_ - 1;
  if (pos >= L_) pos = 0;
  Xc[(size_t)bl * 96 + t] = (_Float16)x[((size_t)b * L_ + pos) * C_ + c];
}

// im2col for a group conv (16 channels, circular padding within the segment).
__global__ void k_im2col_grp(const float* __restrict__ x, const int* __restrict__ cpp,
                             int g, _Float16* __restrict__ Xc) {
  int bl = blockIdx.x; int b = bl / L_, l = bl % L_;
  int t = threadIdx.x;  // 64 threads, cols 48..63 are zero pad (K padded to 64)
  float v = 0.0f;
  if (t < 48) {
    int cp = *cpp;
    int s = (l < cp) ? 0 : cp;
    int e = (l < cp) ? cp : L_;
    int kk = t / 16, c = t % 16;
    int pos = l - 1 + kk;
    if (pos < s)  pos = e - 1;
    if (pos >= e) pos = s;
    v = x[((size_t)b * L_ + pos) * C_ + g * 16 + c];
  }
  Xc[(size_t)bl * 64 + t] = (_Float16)v;
}

// conv-gemm output + sinusoid positional PE (segment-local) + 4 temporal tables.
__global__ __launch_bounds__(256) void k_embed_finish(
    const float* __restrict__ G, const int* __restrict__ xm,
    const int* __restrict__ cpp, int isGlob,
    float* __restrict__ Ef, _Float16* __restrict__ Eh) {
  int bl = blockIdx.x; int l = bl % L_;
  int s = 0;
  if (!isGlob) { int cp = *cpp; s = (l < cp) ? 0 : cp; }
  float pos = (float)(l - s);
  float m0 = (float)xm[bl * 4 + 0], m1 = (float)xm[bl * 4 + 1];
  float m2 = (float)xm[bl * 4 + 2], m3 = (float)xm[bl * 4 + 3];
  for (int d = threadIdx.x; d < D_; d += 256) {
    float i2 = (float)(d & ~1);
    float div = expf(i2 * (-9.210340371976184f / (float)D_));  // ln(10000)
    float v = G[(size_t)bl * D_ + d];
    if (d & 1)
      v += cosf(pos*div) + cosf(m0*div) + cosf(m1*div) + cosf(m2*div) + cosf(m3*div);
    else
      v += sinf(pos*div) + sinf(m0*div) + sinf(m1*div) + sinf(m2*div) + sinf(m3*div);
    Ef[(size_t)bl * D_ + d] = v;
    Eh[(size_t)bl * D_ + d] = (_Float16)v;
  }
}

// Softmax diagonal of mean attention (change-point pass): one block per (b*h, l).
__global__ __launch_bounds__(256) void k_softmax_diag(
    const float* __restrict__ S, float* __restrict__ diag) {
  int l = blockIdx.x, bh = blockIdx.y, t = threadIdx.x;
  const float* row = S + ((size_t)bh * L_ + l) * L_;
  __shared__ float red[256];
  float m = -3.0e38f;
  for (int j = t; j < L_; j += 256) m = fmaxf(m, row[j] * 0.125f);
  red[t] = m; __syncthreads();
  for (int k = 128; k > 0; k >>= 1) { if (t < k) red[t] = fmaxf(red[t], red[t+k]); __syncthreads(); }
  m = red[0]; __syncthreads();
  float sum = 0.0f;
  for (int j = t; j < L_; j += 256) sum += expf(row[j] * 0.125f - m);
  red[t] = sum; __syncthreads();
  for (int k = 128; k > 0; k >>= 1) { if (t < k) red[t] += red[t+k]; __syncthreads(); }
  if (t == 0) {
    float dv = expf(row[l] * 0.125f - m) / red[0];
    atomicAdd(&diag[l], dv * (1.0f / (B_ * H_)));
  }
}

// argmax of |diff(diag)| with NumPy-argsort tie-break (largest index); cp = i+1.
__global__ __launch_bounds__(256) void k_changepoint(
    const float* __restrict__ diag, int* __restrict__ cp) {
  int t = threadIdx.x;
  __shared__ float bv[256]; __shared__ int bi[256];
  float best = -1.0f; int bidx = 0;
  for (int i = t; i < L_ - 1; i += 256) {
    float d = fabsf(diag[i+1] - diag[i]);
    if (d > best || (d == best && i > bidx)) { best = d; bidx = i; }
  }
  bv[t] = best; bi[t] = bidx; __syncthreads();
  for (int k = 128; k > 0; k >>= 1) {
    if (t < k) {
      if (bv[t+k] > bv[t] || (bv[t+k] == bv[t] && bi[t+k] > bi[t])) { bv[t] = bv[t+k]; bi[t] = bi[t+k]; }
    }
    __syncthreads();
  }
  if (t == 0) *cp = bi[0] + 1;
}

// Segment-masked softmax -> f16 probs (zeros outside segment so AV GEMM is unmasked).
__global__ __launch_bounds__(256) void k_softmax_probs(
    const float* __restrict__ S, const int* __restrict__ cpp, _Float16* __restrict__ P) {
  int l = blockIdx.x, bh = blockIdx.y, t = threadIdx.x;
  int cp = *cpp;
  int s = (l < cp) ? 0 : cp;
  int e = (l < cp) ? cp : L_;
  const float* row = S + ((size_t)bh * L_ + l) * L_;
  _Float16* prow   = P + ((size_t)bh * L_ + l) * L_;
  __shared__ float red[256];
  float m = -3.0e38f;
  for (int j = s + t; j < e; j += 256) m = fmaxf(m, row[j] * 0.125f);
  red[t] = m; __syncthreads();
  for (int k = 128; k > 0; k >>= 1) { if (t < k) red[t] = fmaxf(red[t], red[t+k]); __syncthreads(); }
  m = red[0]; __syncthreads();
  float sum = 0.0f;
  for (int j = s + t; j < e; j += 256) sum += expf(row[j] * 0.125f - m);
  red[t] = sum; __syncthreads();
  for (int k = 128; k > 0; k >>= 1) { if (t < k) red[t] += red[t+k]; __syncthreads(); }
  float inv = 1.0f / red[0];
  for (int j = t; j < L_; j += 256)
    prow[j] = (_Float16)((j >= s && j < e) ? expf(row[j] * 0.125f - m) * inv : 0.0f);
}

// y = LN(x + r) * g + b ; writes f32 and (optionally) f16. One block per (b,l).
__global__ __launch_bounds__(256) void k_add_ln(
    const float* __restrict__ X, const float* __restrict__ R,
    const float* __restrict__ g, const float* __restrict__ be,
    float* __restrict__ Y, _Float16* __restrict__ Yh) {
  int bl = blockIdx.x, t = threadIdx.x;
  const float* x = X + (size_t)bl * D_;
  const float* r = R + (size_t)bl * D_;
  float v0 = x[t] + r[t];
  float v1 = x[t + 256] + r[t + 256];
  __shared__ float red[256];
  red[t] = v0 + v1; __syncthreads();
  for (int k = 128; k > 0; k >>= 1) { if (t < k) red[t] += red[t+k]; __syncthreads(); }
  float mu = red[0] * (1.0f / D_); __syncthreads();
  float d0 = v0 - mu, d1 = v1 - mu;
  red[t] = d0 * d0 + d1 * d1; __syncthreads();
  for (int k = 128; k > 0; k >>= 1) { if (t < k) red[t] += red[t+k]; __syncthreads(); }
  float rstd = rsqrtf(red[0] * (1.0f / D_) + 1e-5f);
  float y0 = d0 * rstd * g[t] + be[t];
  float y1 = d1 * rstd * g[t + 256] + be[t + 256];
  Y[(size_t)bl * D_ + t] = y0;
  Y[(size_t)bl * D_ + t + 256] = y1;
  if (Yh) {
    Yh[(size_t)bl * D_ + t] = (_Float16)y0;
    Yh[(size_t)bl * D_ + t + 256] = (_Float16)y1;
  }
}

__global__ void k_zero_f32(float* __restrict__ p, int n) {
  int i = blockIdx.x * 256 + threadIdx.x;
  if (i < n) p[i] = 0.0f;
}

// Final projection: out[b,lp,c] = dot(X2[dir(c)][b,l,:], proj_w) + proj_b.
__global__ __launch_bounds__(256) void k_project(
    const float* __restrict__ X0, const float* __restrict__ X1,
    const float* __restrict__ pw, const float* __restrict__ pb,
    float* __restrict__ out) {
  int bp = blockIdx.x; int b = bp / PRED_, lp = bp % PRED_;
  int l = L_ - PRED_ + lp; int t = threadIdx.x;
  size_t row = ((size_t)b * L_ + l) * D_;
  float s0 = 0.0f, s1 = 0.0f;
  for (int d = t; d < D_; d += 256) { float w = pw[d]; s0 += X0[row + d] * w; s1 += X1[row + d] * w; }
  __shared__ float r0[256], r1[256];
  r0[t] = s0; r1[t] = s1; __syncthreads();
  for (int k = 128; k > 0; k >>= 1) { if (t < k) { r0[t] += r0[t+k]; r1[t] += r1[t+k]; } __syncthreads(); }
  if (t < C_)
    out[(size_t)bp * C_ + t] = ((t < 16) ? r0[0] : r1[0]) + pb[0];
}

// ---------------------------------------------------------------------------
extern "C" void kernel_launch(void* const* d_in, const int* in_sizes, int n_in,
                              void* d_out, int out_size, void* d_ws, size_t ws_size,
                              hipStream_t stream) {
  (void)in_sizes; (void)n_in; (void)out_size; (void)ws_size;
  const float* x_enc  = (const float*)d_in[0];
  const int*   x_mark = (const int*)d_in[1];
  const float* glob_conv = (const float*)d_in[4];
  const float* grp_conv0 = (const float*)d_in[5];
  const float* grp_conv1 = (const float*)d_in[6];
  const float* t_wq = (const float*)d_in[7];
  const float* t_wk = (const float*)d_in[8];
  const float* t_bq = (const float*)d_in[11];
  const float* t_bk = (const float*)d_in[12];
  const float* l_w[2][4]; const float* l_b[2][4];
  for (int g = 0; g < 2; ++g)
    for (int i = 0; i < 4; ++i) {
      l_w[g][i] = (const float*)d_in[15 + 8 * g + i];
      l_b[g][i] = (const float*)d_in[15 + 8 * g + 4 + i];
    }
  const float* n1g = (const float*)d_in[31];
  const float* n1b = (const float*)d_in[32];
  const float* n2g = (const float*)d_in[33];
  const float* n2b = (const float*)d_in[34];
  const float* mlp_w1 = (const float*)d_in[35];
  const float* mlp_b1 = (const float*)d_in[36];
  const float* mlp_w2 = (const float*)d_in[37];
  const float* mlp_b2 = (const float*)d_in[38];
  const float* proj_w = (const float*)d_in[39];
  const float* proj_b = (const float*)d_in[40];

  // ---- workspace carve (256B aligned slabs) ----
  char* w = (char*)d_ws; size_t off = 0;
  auto carve = [&](size_t bytes) -> void* {
    void* p = w + off; off = (off + bytes + 255) & ~(size_t)255; return p;
  };
  const int BL = B_ * L_;
  float*    S    = (float*)carve((size_t)B_*H_*L_*L_*4);    // scores
  _Float16* P    = (_Float16*)carve((size_t)B_*H_*L_*L_*2); // probs f16
  float*    Ef   = (float*)carve((size_t)BL*D_*4);
  _Float16* Eh   = (_Float16*)carve((size_t)BL*D_*2);
  _Float16* Qf   = (_Float16*)carve((size_t)BL*D_*2);
  _Float16* Kf   = (_Float16*)carve((size_t)BL*D_*2);
  _Float16* Vt   = (_Float16*)carve((size_t)BL*D_*2);       // (B,H,dk,L)
  _Float16* AV   = (_Float16*)carve((size_t)BL*D_*2);
  float*    ATT  = (float*)carve((size_t)BL*D_*4);
  float*    X1   = (float*)carve((size_t)BL*D_*4);
  _Float16* X1h  = (_Float16*)carve((size_t)BL*D_*2);
  _Float16* Hff  = (_Float16*)carve((size_t)BL*DFF_*2);
  float*    G32  = (float*)carve((size_t)BL*D_*4);          // gemm f32 scratch
  float*    X2a  = (float*)carve((size_t)BL*D_*4);
  float*    X2b  = (float*)carve((size_t)BL*D_*4);
  _Float16* Xcol = (_Float16*)carve((size_t)BL*96*2);
  _Float16* cw_g = (_Float16*)carve((size_t)D_*96*2);
  _Float16* cw_0 = (_Float16*)carve((size_t)D_*64*2);
  _Float16* cw_1 = (_Float16*)carve((size_t)D_*64*2);
  _Float16* twq_t = (_Float16*)carve((size_t)D_*D_*2);
  _Float16* twk_t = (_Float16*)carve((size_t)D_*D_*2);
  _Float16* lw_t[2][4];
  for (int g = 0; g < 2; ++g)
    for (int i = 0; i < 4; ++i) lw_t[g][i] = (_Float16*)carve((size_t)D_*D_*2);
  _Float16* w1t  = (_Float16*)carve((size_t)DFF_*D_*2);
  _Float16* w2t  = (_Float16*)carve((size_t)D_*DFF_*2);
  float*    diag = (float*)carve((size_t)L_*4);
  int*      cp   = (int*)carve(256);

  // ---- weight prep: f32 -> f16 transposed (N x Kpad) ----
  auto WT = [&](const float* src, _Float16* dst, int K, int N, int Kpad) {
    int tot = N * Kpad;
    k_w_t<<<dim3((tot + 255) / 256), dim3(256), 0, stream>>>(src, dst, K, N, Kpad);
  };
  WT(glob_conv, cw_g, 96, D_, 96);
  WT(grp_conv0, cw_0, 48, D_, 64);
  WT(grp_conv1, cw_1, 48, D_, 64);
  WT(t_wq, twq_t, D_, D_, D_);
  WT(t_wk, twk_t, D_, D_, D_);
  for (int g = 0; g < 2; ++g)
    for (int i = 0; i < 4; ++i) WT(l_w[g][i], lw_t[g][i], D_, D_, D_);
  WT(mlp_w1, w1t, D_, DFF_, D_);
  WT(mlp_w2, w2t, DFF_, D_, DFF_);

  auto GEMM = [&](const _Float16* A, long long sAb, long long sAh, int lda,
                  const _Float16* Bt, long long sBb, long long sBh, int ldbt,
                  float* Cc, long long sCb, long long sCh, int ldc,
                  _Float16* C16, long long sC16b, long long sC16h, int ldc16,
                  _Float16* Tp, const float* bias,
                  int M, int N, int K, int Hdiv, int batch, int act) {
    dim3 grid(N / BN, M / BM, batch);
    k_gemm<<<grid, dim3(256), 0, stream>>>(A, sAb, sAh, lda, Bt, sBb, sBh, ldbt,
        Cc, sCb, sCh, ldc, C16, sC16b, sC16h, ldc16, Tp, bias, M, N, K, Hdiv, act);
  };

  // ================= change-point pass (global attention, Q/K only) =========
  k_im2col_glob<<<dim3(BL), dim3(128), 0, stream>>>(x_enc, Xcol);
  GEMM(Xcol,0,0,96, cw_g,0,0,96, G32,0,0,D_, nullptr,0,0,0, nullptr, nullptr,
       BL, D_, 96, 1, 1, 0);
  k_embed_finish<<<dim3(BL), dim3(256), 0, stream>>>(G32, x_mark, cp, 1, Ef, Eh);
  GEMM(Eh,0,0,D_, twq_t,0,0,D_, nullptr,0,0,0, Qf,0,0,D_, nullptr, t_bq,
       BL, D_, D_, 1, 1, 0);
  GEMM(Eh,0,0,D_, twk_t,0,0,D_, nullptr,0,0,0, Kf,0,0,D_, nullptr, t_bk,
       BL, D_, D_, 1, 1, 0);
  GEMM(Qf, (long long)L_*D_, DK_, D_,
       Kf, (long long)L_*D_, DK_, D_,
       S, (long long)H_*L_*L_, (long long)L_*L_, L_,
       nullptr,0,0,0, nullptr, nullptr,
       L_, L_, DK_, H_, B_*H_, 0);
  k_zero_f32<<<dim3(2), dim3(256), 0, stream>>>(diag, L_);
  k_softmax_diag<<<dim3(L_, B_*H_), dim3(256), 0, stream>>>(S, diag);
  k_changepoint<<<dim3(1), dim3(256), 0, stream>>>(diag, cp);

  // ================= per-group encoder (segments handled via cp mask) =======
  float* X2[2] = {X2a, X2b};
  const _Float16* cwg[2] = {cw_0, cw_1};
  for (int g = 0; g < 2; ++g) {
    k_im2col_grp<<<dim3(BL), dim3(64), 0, stream>>>(x_enc, cp, g, Xcol);
    GEMM(Xcol,0,0,64, cwg[g],0,0,64, G32,0,0,D_, nullptr,0,0,0, nullptr, nullptr,
         BL, D_, 64, 1, 1, 0);
    k_embed_finish<<<dim3(BL), dim3(256), 0, stream>>>(G32, x_mark, cp, 0, Ef, Eh);
    // Q, K, V (V written transposed as (B,H,dk,L))
    GEMM(Eh,0,0,D_, lw_t[g][0],0,0,D_, nullptr,0,0,0, Qf,0,0,D_, nullptr, l_b[g][0],
         BL, D_, D_, 1, 1, 0);
    GEMM(Eh,0,0,D_, lw_t[g][1],0,0,D_, nullptr,0,0,0, Kf,0,0,D_, nullptr, l_b[g][1],
         BL, D_, D_, 1, 1, 0);
    GEMM(Eh,0,0,D_, lw_t[g][2],0,0,D_, nullptr,0,0,0, nullptr,0,0,0, Vt, l_b[g][2],
         BL, D_, D_, 1, 1, 0);
    // scores = Q K^T  (batched over B*H)
    GEMM(Qf, (long long)L_*D_, DK_, D_,
         Kf, (long long)L_*D_, DK_, D_,
         S, (long long)H_*L_*L_, (long long)L_*L_, L_,
         nullptr,0,0,0, nullptr, nullptr,
         L_, L_, DK_, H_, B_*H_, 0);
    k_softmax_probs<<<dim3(L_, B_*H_), dim3(256), 0, stream>>>(S, cp, P);
    // AV (batched): P (L x L) * Vt (dk x L)^T -> AV (B,L,D) f16
    GEMM(P,  (long long)H_*L_*L_, (long long)L_*L_, L_,
         Vt, (long long)H_*DK_*L_, (long long)DK_*L_, L_,
         nullptr,0,0,0,
         AV, (long long)L_*D_, DK_, D_,
         nullptr, nullptr, L_, DK_, L_, H_, B_*H_, 0);
    // output projection of attention
    GEMM(AV,0,0,D_, lw_t[g][3],0,0,D_, ATT,0,0,D_, nullptr,0,0,0, nullptr, l_b[g][3],
         BL, D_, D_, 1, 1, 0);
    k_add_ln<<<dim3(BL), dim3(256), 0, stream>>>(Ef, ATT, n1g, n1b, X1, X1h);
    // MLP: 512 -> 2048 (GELU) -> 512
    GEMM(X1h,0,0,D_, w1t,0,0,D_, nullptr,0,0,0, Hff,0,0,DFF_, nullptr, mlp_b1,
         BL, DFF_, D_, 1, 1, 1);
    GEMM(Hff,0,0,DFF_, w2t,0,0,DFF_, G32,0,0,D_, nullptr,0,0,0, nullptr, mlp_b2,
         BL, D_, DFF_, 1, 1, 0);
    k_add_ln<<<dim3(BL), dim3(256), 0, stream>>>(X1, G32, n2g, n2b, X2[g], nullptr);
  }

  k_project<<<dim3(B_*PRED_), dim3(256), 0, stream>>>(X2[0], X2[1], proj_w, proj_b,
                                                      (float*)d_out);
}
